// XCABlock_80023830659442
// MI455X (gfx1250) — matrix operand order repaired and checked by reading, not compiled, not run on hardware
//
#include <hip/hip_runtime.h>
#include <hip/hip_bf16.h>

// ---------------------------------------------------------------------------
// XCA block for MI455X (gfx1250, wave32, WMMA).
// All GEMMs (QKV / proj / fc1 / fc2 / attention score / attention PV) run on
// v_wmma_f32_16x16x32_f16 with f32 accumulation. Fragment layouts follow
// CDNA5 ISA 7.12.2. Tile staging uses the CDNA5 async copy path
// (global_load_async_to_lds_b128, ASYNCcnt) with double-buffered LDS so the
// next K-step's copies overlap the current K-step's WMMAs.
// ---------------------------------------------------------------------------

typedef __attribute__((ext_vector_type(16))) _Float16 v16h;
typedef __attribute__((ext_vector_type(8)))  float    v8f;

#define B_    32
#define C_    384
#define HW_   1024            // H*W tokens per image
#define NTOK  (B_ * HW_)      // 32768
#define NH_   4
#define HD_   96
#define CC_   192
#define HID_  1536
#define QKVC  (3 * C_)        // 1152

#define WMMA16(a, b, c) \
  __builtin_amdgcn_wmma_f32_16x16x32_f16(false, (a), false, (b), (short)0, (c), false, false)

// CDNA5 async global->LDS copy (16B per lane), tracked by ASYNCcnt.
// GV mode: vdst = LDS byte address (low 32 bits of generic LDS pointer),
// vaddr = 64-bit global address.  (CDNA5 ISA)
__device__ __forceinline__ void async_cp16(void* lds, const void* g) {
  asm volatile("global_load_async_to_lds_b128 %0, %1, off"
               :: "v"((unsigned int)(unsigned long long)lds),
                  "v"((unsigned long long)g)
               : "memory");
}
__device__ __forceinline__ void wait_async0() {
  asm volatile("s_wait_asynccnt 0x0" ::: "memory");
}

// A-matrix fragment, 16x32 f16 (MxK), row-major source with row stride ld
// (halves). Lane L: row = L%16; elements 0-7 = K[(L/16)*8 .. +7],
// elements 8-15 = K[16+(L/16)*8 .. +7]  (ISA 7.12.2, 16-bit A 16x32).
__device__ __forceinline__ v16h frag_a(const _Float16* p, int ld, int lane) {
  const _Float16* row = p + (size_t)(lane & 15) * ld + ((lane >> 4) << 3);
  v16h f;
  ((float4*)&f)[0] = *((const float4*)(row));
  ((float4*)&f)[1] = *((const float4*)(row + 16));
  return f;
}

// B-matrix fragment, 32x16 f16 (KxN). We hold B^T row-major (W[n][k], row
// stride ld halves). Lane L: col n = L%16; elements 0-7 = K[(L/16)*8 .. +7],
// elements 8-15 = K[16+(L/16)*8 .. +7] (same K order as frag_a).
__device__ __forceinline__ v16h frag_b(const _Float16* p, int ld, int lane) {
  const _Float16* row = p + (size_t)(lane & 15) * ld + ((lane >> 4) << 3);
  v16h f;
  ((float4*)&f)[0] = *((const float4*)(row));
  ((float4*)&f)[1] = *((const float4*)(row + 16));
  return f;
}

// ---------------------------------------------------------------------------
// 1) depthwise 3x3 conv on channels [0,192) + passthrough -> shortcut (f32)
// ---------------------------------------------------------------------------
__global__ __launch_bounds__(256) void k_conv_dw(
    const float* __restrict__ x, const float* __restrict__ cw,
    const float* __restrict__ cb, float* __restrict__ s) {
  int bc = blockIdx.x;
  int c = bc % C_;
  const float* xin  = x + (size_t)bc * HW_;
  float*       sout = s + (size_t)bc * HW_;
  if (c < CC_) {
    float w[9];
#pragma unroll
    for (int i = 0; i < 9; ++i) w[i] = cw[c * 9 + i];
    float bias = cb[c];
    for (int px = threadIdx.x; px < HW_; px += 256) {
      int oy = px >> 5, ox = px & 31;
      float acc = bias;
#pragma unroll
      for (int ky = 0; ky < 3; ++ky) {
        int iy = oy + ky - 1;
        if (iy < 0 || iy > 31) continue;
#pragma unroll
        for (int kx = 0; kx < 3; ++kx) {
          int ix = ox + kx - 1;
          if (ix < 0 || ix > 31) continue;
          acc += w[ky * 3 + kx] * xin[iy * 32 + ix];
        }
      }
      sout[px] = acc;
    }
  } else {
    for (int px = threadIdx.x; px < HW_; px += 256) sout[px] = xin[px];
  }
}

// ---------------------------------------------------------------------------
// 2) channel LayerNorm: NCHW f32 in -> token-major [B*N, C] f16 out
// ---------------------------------------------------------------------------
__global__ __launch_bounds__(256) void k_ln(
    const float* __restrict__ x, const float* __restrict__ w,
    const float* __restrict__ bv, _Float16* __restrict__ y) {
  __shared__ float psum[4][64], psq[4][64], rm[64], rs[64];
  int blk = blockIdx.x;            // B_*16 blocks
  int b = blk >> 4, sp0 = (blk & 15) << 6;
  int t = threadIdx.x, sx = t & 63, grp = t >> 6;
  const float* xb = x + (size_t)b * C_ * HW_ + sp0 + sx;
  float s0 = 0.f, s1 = 0.f;
  for (int c = grp; c < C_; c += 4) {
    float v = xb[(size_t)c * HW_];
    s0 += v; s1 += v * v;
  }
  psum[grp][sx] = s0; psq[grp][sx] = s1;
  __syncthreads();
  if (t < 64) {
    float a = psum[0][t] + psum[1][t] + psum[2][t] + psum[3][t];
    float q = psq[0][t] + psq[1][t] + psq[2][t] + psq[3][t];
    float mean = a * (1.f / C_);
    float var  = q * (1.f / C_) - mean * mean;
    rm[t] = mean;
    rs[t] = rsqrtf(var + 1e-6f);
  }
  __syncthreads();
  const float* xg = x + (size_t)b * C_ * HW_ + sp0;
  _Float16*    yg = y + ((size_t)b * HW_ + sp0) * C_;
  for (int idx = t; idx < 64 * C_; idx += 256) {   // coalesced f16 writes
    int sp = idx / C_, c = idx - sp * C_;
    float v = xg[(size_t)c * HW_ + sp];
    yg[idx] = (_Float16)((v - rm[sp]) * rs[sp] * w[c] + bv[c]);
  }
}

// ---------------------------------------------------------------------------
// 3) f32 -> f16 weight conversion
// ---------------------------------------------------------------------------
__global__ void k_cvt(const float* __restrict__ in, _Float16* __restrict__ out, int n) {
  int i = blockIdx.x * 256 + threadIdx.x;
  if (i < n) out[i] = (_Float16)in[i];
}

// ---------------------------------------------------------------------------
// 4) WMMA GEMM: Out[M,Nout] = A[M,K] * W[Nout,K]^T + bias
//    block tile 128x64, 8 waves each 32x32, K-step 32, double-buffered LDS
//    staged with async global->LDS copies (overlaps next tile with WMMAs).
//    MODE 0: f16 out             (qkv)
//    MODE 1: f16 out, exact GELU (fc1)
//    MODE 2: f32 NCHW out = resid + scalevec[c]*(acc+bias)  (proj / fc2)
// ---------------------------------------------------------------------------
template <int K, int MODE>
__global__ __launch_bounds__(256) void k_gemm(
    const _Float16* __restrict__ A, const _Float16* __restrict__ W,
    const float* __restrict__ bias, int Nout,
    const float* __restrict__ scalevec, const float* __restrict__ resid,
    _Float16* __restrict__ out16, float* __restrict__ out32) {
  __shared__ __align__(16) _Float16 As[2][128 * 40];   // pad 32->40 halves
  __shared__ __align__(16) _Float16 Ws[2][64 * 40];
  constexpr int TILES = K / 32;
  int m0 = blockIdx.x * 128, n0 = blockIdx.y * 64;
  int t = threadIdx.x, lane = t & 31, wave = t >> 5;
  int wm = (wave & 3) * 32, wn = (wave >> 2) * 32;

  // per-thread staging coordinates: A rows (t>>2) and (t>>2)+64, W row t>>2,
  // 16-byte chunk (t&3) within the 32-half K slice.
  int rs_ = t >> 2, cs = (t & 3) * 8;
  const _Float16* gA0 = A + (size_t)(m0 + rs_) * K + cs;
  const _Float16* gA1 = A + (size_t)(m0 + rs_ + 64) * K + cs;
  const _Float16* gW  = W + (size_t)(n0 + rs_) * K + cs;

  v8f acc[2][2] = {};
  {
    _Float16* la = &As[0][rs_ * 40 + cs];
    async_cp16(la,            gA0);
    async_cp16(la + 64 * 40,  gA1);
    async_cp16(&Ws[0][rs_ * 40 + cs], gW);
  }
  for (int ki = 0; ki < TILES; ++ki) {
    int buf = ki & 1;
    wait_async0();
    __syncthreads();              // tile ki visible; buf^1 readers retired
    if (ki + 1 < TILES) {
      int ko = (ki + 1) * 32;
      _Float16* la = &As[buf ^ 1][rs_ * 40 + cs];
      async_cp16(la,            gA0 + ko);
      async_cp16(la + 64 * 40,  gA1 + ko);
      async_cp16(&Ws[buf ^ 1][rs_ * 40 + cs], gW + ko);
    }
    v16h a0 = frag_a(&As[buf][wm * 40], 40, lane);
    v16h a1 = frag_a(&As[buf][(wm + 16) * 40], 40, lane);
    v16h b0 = frag_b(&Ws[buf][wn * 40], 40, lane);
    v16h b1 = frag_b(&Ws[buf][(wn + 16) * 40], 40, lane);
    acc[0][0] = WMMA16(a0, b0, acc[0][0]);
    acc[0][1] = WMMA16(a0, b1, acc[0][1]);
    acc[1][0] = WMMA16(a1, b0, acc[1][0]);
    acc[1][1] = WMMA16(a1, b1, acc[1][1]);
  }
  int hi = lane >> 4, ln = lane & 15;
#pragma unroll
  for (int ma = 0; ma < 2; ++ma)
#pragma unroll
    for (int nb = 0; nb < 2; ++nb) {
      int col = n0 + wn + nb * 16 + ln;
      float bb = bias[col];
#pragma unroll
      for (int i = 0; i < 8; ++i) {
        int row = m0 + wm + ma * 16 + hi * 8 + i;   // C layout: row=(L/16)*8+i
        float v = acc[ma][nb][i] + bb;
        if constexpr (MODE == 0) {
          out16[(size_t)row * Nout + col] = (_Float16)v;
        } else if constexpr (MODE == 1) {
          out16[(size_t)row * Nout + col] =
              (_Float16)(0.5f * v * (1.f + erff(v * 0.70710678118654752f)));
        } else {
          int b = row >> 10, sp = row & 1023;       // contiguous f32 stores
          size_t idx = ((size_t)b * C_ + col) * HW_ + sp;
          out32[idx] = resid[idx] + scalevec[col] * v;
        }
      }
    }
}

// ---------------------------------------------------------------------------
// 5) L2-normalize q and k rows (96 elems) in place; one wave per row
// ---------------------------------------------------------------------------
__global__ __launch_bounds__(256) void k_l2norm(_Float16* __restrict__ qkv) {
  int rid  = blockIdx.x * 8 + (threadIdx.x >> 5);
  int lane = threadIdx.x & 31;
  int tok = rid >> 3, rest = rid & 7;
  int which = rest >> 2, head = rest & 3;           // 0=q, 1=k
  _Float16* p = qkv + (size_t)tok * QKVC + which * C_ + head * HD_;
  float vs[3], ss = 0.f;
#pragma unroll
  for (int j = 0; j < 3; ++j) { float v = (float)p[lane + j * 32]; vs[j] = v; ss += v * v; }
#pragma unroll
  for (int off = 16; off > 0; off >>= 1) ss += __shfl_xor(ss, off);
  float sc = 1.f / fmaxf(sqrtf(ss), 1e-12f);
#pragma unroll
  for (int j = 0; j < 3; ++j) p[lane + j * 32] = (_Float16)(vs[j] * sc);
}

// ---------------------------------------------------------------------------
// 6) attention (flash-style): block = 128 queries of one (b, head);
//    8 waves x 16 query rows. Q frags in registers; K chunk async-copied to
//    LDS row-major, V chunk transposed in LDS (PV B-frags contiguous);
//    online softmax with width-16 shfl_xor butterflies (matches C-matrix
//    lane layout); P converted C->A layout through per-wave LDS scratch.
// ---------------------------------------------------------------------------
__global__ __launch_bounds__(256) void k_attn(
    const _Float16* __restrict__ qkv, const float* __restrict__ temp,
    _Float16* __restrict__ o16) {
  __shared__ __align__(16) _Float16 Ks[64 * 104];   // 64 keys x 96(+8) d
  __shared__ __align__(16) _Float16 Vt[96 * 72];    // 96 d x 64(+8) keys
  __shared__ __align__(16) _Float16 Ps[8 * 16 * 72];
  int blk = blockIdx.x;                             // B*NH*8
  int q0   = (blk & 7) << 7;
  int head = (blk >> 3) & 3;
  int b    = blk >> 5;
  int t = threadIdx.x, lane = t & 31, wave = t >> 5;
  int hi = lane >> 4, ln = lane & 15;
  float scale = temp[head];

  const _Float16* qg = qkv + ((size_t)b * HW_ + q0 + wave * 16) * QKVC + head * HD_;
  v16h qa[3];
#pragma unroll
  for (int kk = 0; kk < 3; ++kk) qa[kk] = frag_a(qg + kk * 32, QKVC, lane);

  v8f  o[6] = {};
  float mrow[8], lrow[8];
#pragma unroll
  for (int i = 0; i < 8; ++i) { mrow[i] = -1e30f; lrow[i] = 0.f; }

  const _Float16* kg = qkv + (size_t)b * HW_ * QKVC + C_     + head * HD_;
  const _Float16* vg = qkv + (size_t)b * HW_ * QKVC + 2 * C_ + head * HD_;
  _Float16* Pw = Ps + (size_t)wave * 16 * 72;

  for (int j0 = 0; j0 < HW_; j0 += 64) {
#pragma unroll
    for (int f = t; f < 768; f += 256) {            // stage K: async -> LDS
      int r = f / 12, q = f % 12;
      async_cp16(Ks + (size_t)r * 104 + q * 8,
                 kg + (size_t)(j0 + r) * QKVC + q * 8);
    }
#pragma unroll
    for (int f = t; f < 768; f += 256) {            // stage V transposed
      int r = f / 12, q = f % 12;
      float4 tmp = *((const float4*)(vg + (size_t)(j0 + r) * QKVC + q * 8));
      const _Float16* th = (const _Float16*)&tmp;
#pragma unroll
      for (int e = 0; e < 8; ++e) Vt[(size_t)(q * 8 + e) * 72 + r] = th[e];
    }
    wait_async0();
    __syncthreads();

    v8f s[4] = {};                                   // S = Q*K^T (16x64)
#pragma unroll
    for (int ns = 0; ns < 4; ++ns)
#pragma unroll
      for (int kk = 0; kk < 3; ++kk)
        s[ns] = WMMA16(qa[kk], frag_b(Ks + (size_t)(ns * 16) * 104 + kk * 32, 104, lane), s[ns]);

#pragma unroll
    for (int ns = 0; ns < 4; ++ns)
#pragma unroll
      for (int i = 0; i < 8; ++i) s[ns][i] *= scale;

#pragma unroll
    for (int i = 0; i < 8; ++i) {                    // online softmax per row
      float v = fmaxf(fmaxf(s[0][i], s[1][i]), fmaxf(s[2][i], s[3][i]));
#pragma unroll
      for (int off = 1; off < 16; off <<= 1) v = fmaxf(v, __shfl_xor(v, off));
      float mn = fmaxf(mrow[i], v);
      float corr = __expf(mrow[i] - mn);
      mrow[i] = mn;
      float rsum = 0.f;
#pragma unroll
      for (int ns = 0; ns < 4; ++ns) {
        float p = __expf(s[ns][i] - mn);
        s[ns][i] = p;
        rsum += p;
      }
#pragma unroll
      for (int off = 1; off < 16; off <<= 1) rsum += __shfl_xor(rsum, off);
      lrow[i] = lrow[i] * corr + rsum;
#pragma unroll
      for (int nd = 0; nd < 6; ++nd) o[nd][i] *= corr;
    }

#pragma unroll
    for (int ns = 0; ns < 4; ++ns)                   // P: C layout -> LDS
#pragma unroll
      for (int i = 0; i < 8; ++i)
        Pw[(size_t)(hi * 8 + i) * 72 + ns * 16 + ln] = (_Float16)s[ns][i];

    v16h pa0 = frag_a(Pw, 72, lane);                 // O += P*V
    v16h pa1 = frag_a(Pw + 32, 72, lane);
#pragma unroll
    for (int nd = 0; nd < 6; ++nd) {
      o[nd] = WMMA16(pa0, frag_b(Vt + (size_t)(nd * 16) * 72,      72, lane), o[nd]);
      o[nd] = WMMA16(pa1, frag_b(Vt + (size_t)(nd * 16) * 72 + 32, 72, lane), o[nd]);
    }
    __syncthreads();                                 // retire readers
  }

  _Float16* ob = o16 + ((size_t)b * HW_ + q0 + wave * 16) * C_ + head * HD_;
#pragma unroll
  for (int i = 0; i < 8; ++i) {
    float inv = 1.f / lrow[i];
#pragma unroll
    for (int nd = 0; nd < 6; ++nd)
      ob[(size_t)(hi * 8 + i) * C_ + nd * 16 + ln] = (_Float16)(o[nd][i] * inv);
  }
}

// ---------------------------------------------------------------------------
extern "C" void kernel_launch(void* const* d_in, const int* in_sizes, int n_in,
                              void* d_out, int out_size, void* d_ws, size_t ws_size,
                              hipStream_t stream) {
  (void)in_sizes; (void)n_in; (void)out_size; (void)ws_size;
  const float* x        = (const float*)d_in[0];
  const float* conv_w   = (const float*)d_in[1];
  const float* conv_b   = (const float*)d_in[2];
  const float* nxca_w   = (const float*)d_in[3];
  const float* nxca_b   = (const float*)d_in[4];
  const float* temp     = (const float*)d_in[5];
  const float* qkv_w    = (const float*)d_in[6];
  const float* qkv_b    = (const float*)d_in[7];
  const float* proj_w   = (const float*)d_in[8];
  const float* proj_b   = (const float*)d_in[9];
  const float* norm_w   = (const float*)d_in[10];
  const float* norm_b   = (const float*)d_in[11];
  const float* fc1_w    = (const float*)d_in[12];
  const float* fc1_b    = (const float*)d_in[13];
  const float* fc2_w    = (const float*)d_in[14];
  const float* fc2_b    = (const float*)d_in[15];
  const float* gamma_x  = (const float*)d_in[16];
  const float* gamma    = (const float*)d_in[17];
  float* out = (float*)d_out;

  char* ws = (char*)d_ws;
  size_t off = 0;
  float*    s      = (float*)(ws + off);    off += (size_t)B_ * C_ * HW_ * 4;
  _Float16* y16    = (_Float16*)(ws + off); off += (size_t)NTOK * C_ * 2;   // also z16
  _Float16* qkv16  = (_Float16*)(ws + off); off += (size_t)NTOK * QKVC * 2;
  _Float16* o16    = (_Float16*)(ws + off); off += (size_t)NTOK * C_ * 2;
  _Float16* h16    = (_Float16*)(ws + off); off += (size_t)NTOK * HID_ * 2;
  _Float16* qkvw16 = (_Float16*)(ws + off); off += (size_t)QKVC * C_ * 2;
  _Float16* projw16= (_Float16*)(ws + off); off += (size_t)C_ * C_ * 2;
  _Float16* fc1w16 = (_Float16*)(ws + off); off += (size_t)HID_ * C_ * 2;
  _Float16* fc2w16 = (_Float16*)(ws + off); off += (size_t)C_ * HID_ * 2;

  // weight conversions
  k_cvt<<<(QKVC * C_ + 255) / 256, 256, 0, stream>>>(qkv_w,  qkvw16,  QKVC * C_);
  k_cvt<<<(C_ * C_   + 255) / 256, 256, 0, stream>>>(proj_w, projw16, C_ * C_);
  k_cvt<<<(HID_ * C_ + 255) / 256, 256, 0, stream>>>(fc1_w,  fc1w16,  HID_ * C_);
  k_cvt<<<(C_ * HID_ + 255) / 256, 256, 0, stream>>>(fc2_w,  fc2w16,  C_ * HID_);

  // 1) depthwise conv + passthrough -> shortcut
  k_conv_dw<<<B_ * C_, 256, 0, stream>>>(x, conv_w, conv_b, s);
  // 2) LN -> y16 (token-major f16)
  k_ln<<<B_ * 16, 256, 0, stream>>>(s, nxca_w, nxca_b, y16);
  // 3) QKV GEMM
  k_gemm<C_, 0><<<dim3(NTOK / 128, QKVC / 64), 256, 0, stream>>>(
      y16, qkvw16, qkv_b, QKVC, nullptr, nullptr, qkv16, nullptr);
  // 4) q/k L2 normalize
  k_l2norm<<<NTOK, 256, 0, stream>>>(qkv16);
  // 5) attention -> o16
  k_attn<<<B_ * NH_ * 8, 256, 0, stream>>>(qkv16, temp, o16);
  // 6) proj GEMM + residual: d_out = shortcut + gamma_xca * proj(o)
  k_gemm<C_, 2><<<dim3(NTOK / 128, C_ / 64), 256, 0, stream>>>(
      o16, projw16, proj_b, C_, gamma_x, s, nullptr, out);
  // 7) LN2 -> z16 (reuse y16)
  k_ln<<<B_ * 16, 256, 0, stream>>>(out, norm_w, norm_b, y16);
  // 8) fc1 + GELU -> h16
  k_gemm<C_, 1><<<dim3(NTOK / 128, HID_ / 64), 256, 0, stream>>>(
      y16, fc1w16, fc1_b, HID_, nullptr, nullptr, h16, nullptr);
  // 9) fc2 + residual (in place on d_out)
  k_gemm<HID_, 2><<<dim3(NTOK / 128, C_ / 64), 256, 0, stream>>>(
      h16, fc2w16, fc2_b, C_, gamma, out, nullptr, out);
}
